// PairUpdate_52810917871759
// MI455X (gfx1250) — compile-verified, matrix operand order checked by reading, not run on hardware
//
#include <hip/hip_runtime.h>

// ---------------------------------------------------------------------------
// Types
// ---------------------------------------------------------------------------
typedef __attribute__((ext_vector_type(16))) _Float16 v16h;
typedef __attribute__((ext_vector_type(8)))  _Float16 v8h;
typedef __attribute__((ext_vector_type(8)))  float    v8f;

#define LL 256          // seq len
#define EE 128          // embedding dim
#define NPOS (LL * LL)  // 65536 pair positions

// GEMM epilogue flags
enum {
  F_SIG      = 1,    // sigmoid(t)
  F_GATE     = 2,    // t *= gate[row, col]
  F_MASKPRE  = 4,    // t *= mask[row]   (before residual)
  F_RES      = 8,    // t += res[row, col]
  F_MASKPOST = 16,   // t *= mask[row]   (after residual)
  F_F16      = 32,   // store f16 row-major
  F_TRN      = 64,   // store f16 channel-major  out[col][row]      ('in' mode)
  F_TRNOUT   = 128,  // store f16 channel-major  out[col][j][k]     ('out' mode)
};

// ---------------------------------------------------------------------------
// WMMA helpers (CDNA5 16x16x32 f16 -> f32)
// ---------------------------------------------------------------------------
static __device__ __forceinline__ v8f wmma_f16(v16h a, v16h b, v8f c) {
  return __builtin_amdgcn_wmma_f32_16x16x32_f16(false, a, false, b, (short)0, c,
                                                false, false);
}

// Load one 16x32 A-frag (or mirrored 32x16 B-frag) from a row-major slab.
// p points at (row-of-this-lane, k-chunk base). ISA layout: lane&15 = M/N,
// lane bit4 selects K offset {0..7,16..23} vs {8..15,24..31}.
static __device__ __forceinline__ v16h load_frag16(const _Float16* p, int lane) {
  const int kb = (lane & 16) ? 8 : 0;
  v8h lo = *(const v8h*)(p + kb);
  v8h hi = *(const v8h*)(p + kb + 16);
  v16h r;
#pragma unroll
  for (int i = 0; i < 8; ++i) { r[i] = lo[i]; r[8 + i] = hi[i]; }
  return r;
}

// ---------------------------------------------------------------------------
// Generic fused GEMM:  out = epilogue( A(f16 [N,K]) @ W(f32 [K,Dout]) )
// Weight is converted f32->f16 into LDS, transposed to [n][k] so B-frags are
// contiguous ds_load_b128. 8 waves/block; each wave owns a 16x64 tile
// (4 accumulators) so the A-fragment is reused across 4 WMMAs.
// ---------------------------------------------------------------------------
__global__ __launch_bounds__(256)
void gemm16_kernel(const _Float16* __restrict__ A,
                   const float* __restrict__ W,
                   const float* __restrict__ bias,
                   const float* __restrict__ gate,
                   const float* __restrict__ maskv,
                   const float* __restrict__ res,
                   void* __restrict__ out,
                   int nrows, int K, int dout, float scale, int flags) {
  extern __shared__ _Float16 Wt[];  // [dout][K]
  const int tid = threadIdx.x;
  const int total = K * dout;
  const int sh = 31 - __clz(dout);  // dout is a power of two
  for (int e = tid; e < total; e += 256) {
    int k = e >> sh, n = e & (dout - 1);
    Wt[n * K + k] = (_Float16)W[e];
  }
  __syncthreads();

  const int wave = tid >> 5;
  const int lane = tid & 31;
  const int l15  = lane & 15;
  const int tiles_n = dout >> 6;  // 64-wide column tiles
  const int tile = blockIdx.x * 8 + wave;
  const int tm = tile / tiles_n;
  const int tn = tile - tm * tiles_n;

  const _Float16* arow = A + (size_t)(tm * 16 + l15) * K;

  v8f acc[4] = {};
  for (int kk = 0; kk < K; kk += 32) {
    v16h af = load_frag16(arow + kk, lane);
#pragma unroll
    for (int j = 0; j < 4; ++j) {
      v16h bf = load_frag16(&Wt[(size_t)(tn * 64 + j * 16 + l15) * K + kk], lane);
      acc[j] = wmma_f16(af, bf, acc[j]);
    }
  }

  const int hi = (lane & 16) ? 8 : 0;
#pragma unroll
  for (int j = 0; j < 4; ++j) {
#pragma unroll
    for (int r = 0; r < 8; ++r) {
      int row = tm * 16 + hi + r;
      int col = tn * 64 + j * 16 + l15;
      float t = acc[j][r] * scale + (bias ? bias[col] : 0.0f);
      if (flags & F_SIG)      t = 1.0f / (1.0f + __expf(-t));
      if (flags & F_GATE)     t *= gate[(size_t)row * dout + col];
      if (flags & F_MASKPRE)  t *= maskv[row];
      if (flags & F_RES)      t += res[(size_t)row * dout + col];
      if (flags & F_MASKPOST) t *= maskv[row];
      if (flags & (F_TRN | F_TRNOUT)) {
        // channel-major store for the triangular multiply
        size_t idx = (flags & F_TRNOUT)
            ? ((size_t)col * nrows + (size_t)(row & (LL - 1)) * LL + (row >> 8))
            : ((size_t)col * nrows + row);
        ((_Float16*)out)[idx] = (_Float16)t;
      } else if (flags & F_F16) {
        ((_Float16*)out)[(size_t)row * dout + col] = (_Float16)t;
      } else {
        ((float*)out)[(size_t)row * dout + col] = t;
      }
    }
  }
}

// ---------------------------------------------------------------------------
// Triangular multiply: U[i,j,c] = sum_k P[c][i][k] * Q[c][j][k]
// P, Q are channel-major f16 [CU][L][L]; U is [L*L, CU] f32.
// Each wave owns a 16x64 tile (A-frag reused across 4 WMMAs per k-step).
// ---------------------------------------------------------------------------
__global__ __launch_bounds__(256)
void trimul_kernel(const _Float16* __restrict__ P,
                   const _Float16* __restrict__ Q,
                   float* __restrict__ U) {
  const int wave = threadIdx.x >> 5;
  const int lane = threadIdx.x & 31;
  const int l15  = lane & 15;
  const int tile = blockIdx.x * 8 + wave;  // CU * 16 * 4 tiles
  const int c    = tile >> 6;
  const int rem  = tile & 63;
  const int ti = rem >> 2, tj4 = rem & 3;

  const _Float16* arow = P + ((size_t)c << 16) + (size_t)(ti * 16 + l15) * LL;
  const _Float16* bbase = Q + ((size_t)c << 16) + (size_t)(tj4 * 64 + l15) * LL;

  v8f acc[4] = {};
  for (int kk = 0; kk < LL; kk += 32) {
    v16h af = load_frag16(arow + kk, lane);
#pragma unroll
    for (int j = 0; j < 4; ++j) {
      v16h bf = load_frag16(bbase + (size_t)j * 16 * LL + kk, lane);
      acc[j] = wmma_f16(af, bf, acc[j]);
    }
  }
  const int hi = (lane & 16) ? 8 : 0;
#pragma unroll
  for (int j = 0; j < 4; ++j) {
#pragma unroll
    for (int r = 0; r < 8; ++r) {
      int i = ti * 16 + hi + r;
      int jj = tj4 * 64 + j * 16 + l15;
      U[((size_t)i * LL + jj) * 128 + c] = acc[j][r];
    }
  }
}

// ---------------------------------------------------------------------------
// Flash attention: one wave per (s, h, 16-query tile); 8 waves of a block
// share (s, h), so Q/V key-tiles are staged cooperatively in LDS.
// logits[q,v] = K[q,:].Q[v,:] + pbc[h][q,v]   (pbc = 1e9*(mask-1) + pair bias)
// out[q,c] = softmax_v @ V, gated by fu; f16 output [N, 256].
// ---------------------------------------------------------------------------
__global__ __launch_bounds__(256)
void triattn_kernel(const _Float16* __restrict__ Qm,
                    const _Float16* __restrict__ Km,
                    const _Float16* __restrict__ Vm,
                    const float* __restrict__ fu,
                    const float* __restrict__ pbc,
                    _Float16* __restrict__ O) {
  __shared__ _Float16 Qs[32][64];     // [v - vt][c]
  __shared__ _Float16 Vs[64][32];     // [c][v - vt]  (transposed for B-frags)
  __shared__ _Float16 Pl[8][16][32];  // per-wave P staging (C- -> A-layout)
  const int tid  = threadIdx.x;
  const int wave = tid >> 5;
  const int lane = tid & 31;
  const int l15  = lane & 15;
  const int hi   = (lane & 16) ? 8 : 0;
  const int s = blockIdx.z, h = blockIdx.y;
  const int q0 = (blockIdx.x * 8 + wave) * 16;

  const _Float16* krow = Km + (size_t)(s * LL + q0 + l15) * 256 + h * 64;
  v16h ak0 = load_frag16(krow, lane);
  v16h ak1 = load_frag16(krow + 32, lane);

  v8f o[4] = {};
  float mrun[8], lrun[8];
#pragma unroll
  for (int r = 0; r < 8; ++r) { mrun[r] = -1e30f; lrun[r] = 0.0f; }

  for (int vt = 0; vt < LL; vt += 32) {
    __syncthreads();  // previous iteration's Qs/Vs reads complete
    for (int e = tid; e < 2048; e += 256) {
      int v = e >> 6, c = e & 63;
      _Float16 qv = Qm[(size_t)(s * LL + vt + v) * 256 + h * 64 + c];
      _Float16 vv = Vm[(size_t)(s * LL + vt + v) * 256 + h * 64 + c];
      Qs[v][c] = qv;
      Vs[c][v] = vv;
    }
    __syncthreads();

    float sv0[8], sv1[8];
    {
      v8f acc = {};
      acc = wmma_f16(ak0, load_frag16(&Qs[l15][0], lane), acc);
      acc = wmma_f16(ak1, load_frag16(&Qs[l15][32], lane), acc);
#pragma unroll
      for (int r = 0; r < 8; ++r) {
        int q = q0 + hi + r, v = vt + l15;
        sv0[r] = acc[r] + pbc[(size_t)h * NPOS + q * LL + v];
      }
    }
    {
      v8f acc = {};
      acc = wmma_f16(ak0, load_frag16(&Qs[16 + l15][0], lane), acc);
      acc = wmma_f16(ak1, load_frag16(&Qs[16 + l15][32], lane), acc);
#pragma unroll
      for (int r = 0; r < 8; ++r) {
        int q = q0 + hi + r, v = vt + 16 + l15;
        sv1[r] = acc[r] + pbc[(size_t)h * NPOS + q * LL + v];
      }
    }
    // online softmax update per row (rows split across half-waves)
#pragma unroll
    for (int r = 0; r < 8; ++r) {
      float cm = fmaxf(sv0[r], sv1[r]);
#pragma unroll
      for (int m = 8; m >= 1; m >>= 1) cm = fmaxf(cm, __shfl_xor(cm, m, 32));
      float mn = fmaxf(mrun[r], cm);
      float alpha = __expf(mrun[r] - mn);
      float p0 = __expf(sv0[r] - mn);
      float p1 = __expf(sv1[r] - mn);
      float ls = p0 + p1;
#pragma unroll
      for (int m = 8; m >= 1; m >>= 1) ls += __shfl_xor(ls, m, 32);
      lrun[r] = lrun[r] * alpha + ls;
      mrun[r] = mn;
      o[0][r] *= alpha; o[1][r] *= alpha; o[2][r] *= alpha; o[3][r] *= alpha;
      Pl[wave][hi + r][l15]      = (_Float16)p0;
      Pl[wave][hi + r][16 + l15] = (_Float16)p1;
    }
    asm volatile("s_wait_dscnt 0x0" ::: "memory");
    v16h ap = load_frag16(&Pl[wave][l15][0], lane);
#pragma unroll
    for (int j = 0; j < 4; ++j) {
      v16h bv = load_frag16(&Vs[j * 16 + l15][0], lane);
      o[j] = wmma_f16(ap, bv, o[j]);
    }
  }
#pragma unroll
  for (int j = 0; j < 4; ++j) {
#pragma unroll
    for (int r = 0; r < 8; ++r) {
      int q = q0 + hi + r;
      int c = h * 64 + j * 16 + l15;
      size_t idx = (size_t)(s * LL + q) * 256 + c;
      O[idx] = (_Float16)(o[j][r] / lrun[r] * fu[idx]);
    }
  }
}

// ---------------------------------------------------------------------------
// LayerNorm over 128 channels, one wave per row, f16 output.
// ---------------------------------------------------------------------------
__global__ __launch_bounds__(256)
void ln128_kernel(const float* __restrict__ x, const float* __restrict__ g,
                  const float* __restrict__ b, _Float16* __restrict__ y,
                  int nrows) {
  const int wave = threadIdx.x >> 5, lane = threadIdx.x & 31;
  const int row = blockIdx.x * 8 + wave;
  if (row >= nrows) return;
  const float4 v = *(const float4*)(x + (size_t)row * 128 + lane * 4);
  float s = v.x + v.y + v.z + v.w;
#pragma unroll
  for (int m = 16; m >= 1; m >>= 1) s += __shfl_xor(s, m, 32);
  const float mean = s * (1.0f / 128.0f);
  const float d0 = v.x - mean, d1 = v.y - mean, d2 = v.z - mean, d3 = v.w - mean;
  float q = d0 * d0 + d1 * d1 + d2 * d2 + d3 * d3;
#pragma unroll
  for (int m = 16; m >= 1; m >>= 1) q += __shfl_xor(q, m, 32);
  const float inv = rsqrtf(q * (1.0f / 128.0f) + 1e-5f);
  const int c = lane * 4;
  _Float16* yr = y + (size_t)row * 128 + c;
  yr[0] = (_Float16)(d0 * inv * g[c + 0] + b[c + 0]);
  yr[1] = (_Float16)(d1 * inv * g[c + 1] + b[c + 1]);
  yr[2] = (_Float16)(d2 * inv * g[c + 2] + b[c + 2]);
  yr[3] = (_Float16)(d3 * inv * g[c + 3] + b[c + 3]);
}

// ---------------------------------------------------------------------------
// Small helpers
// ---------------------------------------------------------------------------
__global__ void cvt_f16_kernel(const float* __restrict__ in,
                               _Float16* __restrict__ out) {
  size_t t = (size_t)blockIdx.x * 256 + threadIdx.x;
  out[t] = (_Float16)in[t];
}

__global__ void transpose_pair_kernel(const float* __restrict__ in,
                                      float* __restrict__ out) {
  size_t t = (size_t)blockIdx.x * 256 + threadIdx.x;  // NPOS*128 threads
  int e = (int)(t & 127);
  size_t ij = t >> 7;
  int j = (int)(ij & (LL - 1));
  int i = (int)(ij >> 8);
  out[(((size_t)j << 8) + i) * 128 + e] = in[t];
}

__global__ void transpose_mask_kernel(const float* __restrict__ in,
                                      float* __restrict__ out) {
  int t = blockIdx.x * 256 + threadIdx.x;  // NPOS threads
  int j = t & (LL - 1), i = t >> 8;
  out[(size_t)j * LL + i] = in[t];
}

// pbc[h][row] = xln[row,:] @ wb[:,h] + 1e9*(mask[row]-1)    (Dout=4 projection)
__global__ void pbias_kernel(const _Float16* __restrict__ xln,
                             const float* __restrict__ wb,
                             const float* __restrict__ ms,
                             float* __restrict__ pbc) {
  int row = blockIdx.x * 256 + threadIdx.x;  // NPOS threads
  float a0 = 0, a1 = 0, a2 = 0, a3 = 0;
  const _Float16* xr = xln + (size_t)row * 128;
#pragma unroll 4
  for (int c = 0; c < 128; ++c) {
    float xv = (float)xr[c];
    const float* w = wb + c * 4;
    a0 += xv * w[0]; a1 += xv * w[1]; a2 += xv * w[2]; a3 += xv * w[3];
  }
  float mb = 1e9f * (ms[row] - 1.0f);
  pbc[0 * (size_t)NPOS + row] = a0 + mb;
  pbc[1 * (size_t)NPOS + row] = a1 + mb;
  pbc[2 * (size_t)NPOS + row] = a2 + mb;
  pbc[3 * (size_t)NPOS + row] = a3 + mb;
}

// ---------------------------------------------------------------------------
// Host orchestration
// ---------------------------------------------------------------------------
extern "C" void kernel_launch(void* const* d_in, const int* in_sizes, int n_in,
                              void* d_out, int out_size, void* d_ws,
                              size_t ws_size, hipStream_t stream) {
  (void)in_sizes; (void)n_in; (void)out_size; (void)ws_size;
  const size_t N = NPOS;
  const float* pair = (const float*)d_in[0];
  const float* mask = (const float*)d_in[1];
  auto P = [&](int i) { return (const float*)d_in[i]; };
  const int TU_IN = 2, TU_OUT = 18, TA_IN = 34, TA_OUT = 44, TR = 54;

  char* ws = (char*)d_ws;
  size_t off = 0;
  auto alloc = [&](size_t bytes) -> void* {
    void* p = ws + off;
    off += (bytes + 255) & ~(size_t)255;
    return p;
  };
  float*    X0  = (float*)alloc(N * 128 * 4);
  float*    X1  = (float*)alloc(N * 128 * 4);
  float*    XT  = (float*)alloc(N * 128 * 4);
  _Float16* XLN = (_Float16*)alloc(N * 128 * 2);
  float*    T1  = (float*)alloc(N * 128 * 4);   // le/re temp; aliased as U
  float*    TU  = T1;                           // einsum output / y_t buffer
  _Float16* LUs = (_Float16*)alloc(N * 128 * 2);
  _Float16* RUs = (_Float16*)alloc(N * 128 * 2);
  float*    TF  = (float*)alloc(N * 256 * 4);   // sigmoid gates (fu)
  _Float16* QH  = (_Float16*)alloc(N * 256 * 2);
  _Float16* KH  = (_Float16*)alloc(N * 256 * 2);
  _Float16* VH  = (_Float16*)alloc(N * 256 * 2);
  _Float16* AH  = (_Float16*)alloc(N * 256 * 2);
  float*    PBC = (float*)alloc(N * 4 * 4);     // [H][N] softmax bias planes
  float*    MT  = (float*)alloc(N * 4);
  _Float16* HH  = QH;  // transition hidden [N,512] aliases QH+KH (contiguous)

  auto gemm = [&](const _Float16* A, const float* W, const float* bias,
                  const float* gate, const float* mv, const float* res,
                  void* out, int K, int Dout, float scale, int flags) {
    int blocks = (int)(N / 16) * (Dout / 64) / 8;
    size_t shm = (size_t)K * Dout * 2;
    gemm16_kernel<<<blocks, 256, shm, stream>>>(A, W, bias, gate, mv, res, out,
                                                (int)N, K, Dout, scale, flags);
  };
  auto ln = [&](const float* x, const float* g, const float* b, _Float16* y) {
    ln128_kernel<<<(int)(N / 8), 256, 0, stream>>>(x, g, b, y, (int)N);
  };

  // ---- triangular multiplicative update ----
  auto tri_update = [&](const float* xin, int b, bool modeOut, float* xout) {
    int trn = F_TRN | (modeOut ? F_TRNOUT : 0);
    ln(xin, P(b + 0), P(b + 1), XLN);
    gemm(XLN, P(b + 2), P(b + 3), nullptr, mask, nullptr, T1, 128, 128, 1.0f,
         F_MASKPRE);                                              // le
    gemm(XLN, P(b + 6), P(b + 7), T1, nullptr, nullptr, LUs, 128, 128, 1.0f,
         F_SIG | F_GATE | trn);                                   // lu (chan-major)
    gemm(XLN, P(b + 4), P(b + 5), nullptr, mask, nullptr, T1, 128, 128, 1.0f,
         F_MASKPRE);                                              // re
    gemm(XLN, P(b + 8), P(b + 9), T1, nullptr, nullptr, RUs, 128, 128, 1.0f,
         F_SIG | F_GATE | trn);                                   // ru (chan-major)
    gemm(XLN, P(b + 10), P(b + 11), nullptr, nullptr, nullptr, TF, 128, 128,
         1.0f, F_SIG);                                            // fu gate
    trimul_kernel<<<128 * 64 / 8, 256, 0, stream>>>(
        modeOut ? RUs : LUs, modeOut ? LUs : RUs, TU);            // einsum
    ln(TU, P(b + 12), P(b + 13), XLN);                            // center LN
    gemm(XLN, P(b + 14), P(b + 15), TF, mask, xin, xout, 128, 128, 1.0f,
         F_GATE | F_MASKPRE | F_RES);                             // proj+gate+res
  };

  // ---- triangular attention ----
  auto tri_attn = [&](const float* xin, int b, bool modeOut, float* xout) {
    const float* xs = xin;
    const float* ms = mask;
    if (modeOut) {
      transpose_pair_kernel<<<(int)(N * 128 / 256), 256, 0, stream>>>(xin, XT);
      transpose_mask_kernel<<<(int)(N / 256), 256, 0, stream>>>(mask, MT);
      xs = XT; ms = MT;
    }
    ln(xs, P(b + 0), P(b + 1), XLN);
    gemm(XLN, P(b + 2), nullptr, nullptr, nullptr, nullptr, QH, 128, 256,
         0.125f, F_F16);                                          // q * CA^-0.5
    gemm(XLN, P(b + 3), nullptr, nullptr, nullptr, nullptr, KH, 128, 256, 1.0f,
         F_F16);                                                  // k
    gemm(XLN, P(b + 4), nullptr, nullptr, nullptr, nullptr, VH, 128, 256, 1.0f,
         F_F16);                                                  // v
    gemm(XLN, P(b + 6), P(b + 7), nullptr, nullptr, nullptr, TF, 128, 256, 1.0f,
         F_SIG);                                                  // fu gate
    pbias_kernel<<<(int)(N / 256), 256, 0, stream>>>(XLN, P(b + 5), ms, PBC);
    triattn_kernel<<<dim3(2, 4, LL), 256, 0, stream>>>(QH, KH, VH, TF, PBC, AH);
    float* dst = modeOut ? TU : xout;
    gemm(AH, P(b + 8), P(b + 9), nullptr, ms, xs, dst, 256, 128, 1.0f,
         F_MASKPRE | F_RES);                                      // out proj+res
    if (modeOut)
      transpose_pair_kernel<<<(int)(N * 128 / 256), 256, 0, stream>>>(TU, xout);
  };

  // ---- pipeline ----
  tri_update(pair, TU_IN, false, X0);
  tri_update(X0, TU_OUT, true, X1);
  tri_attn(X1, TA_IN, false, X0);
  tri_attn(X0, TA_OUT, true, X1);

  // transition MLP: sigmoid(x@w1+b1)@w2 + b2 + x, then final mask
  cvt_f16_kernel<<<(int)(N * 128 / 256), 256, 0, stream>>>(X1, XLN);
  gemm(XLN, P(TR + 0), P(TR + 1), nullptr, nullptr, nullptr, HH, 128, 512, 1.0f,
       F_SIG | F_F16);
  gemm(HH, P(TR + 2), P(TR + 3), nullptr, mask, X1, d_out, 512, 128, 1.0f,
       F_RES | F_MASKPOST);
}
